// GemmaAttention_71940702208200
// MI455X (gfx1250) — matrix operand order repaired and checked by reading, not compiled, not run on hardware
//
#include <hip/hip_runtime.h>
#include <hip/hip_fp16.h>
#include <stdint.h>

typedef _Float16 f16;
typedef _Float16 v16h __attribute__((ext_vector_type(16)));
typedef _Float16 v8h  __attribute__((ext_vector_type(8)));
typedef float    v8f  __attribute__((ext_vector_type(8)));
typedef float    v4f  __attribute__((ext_vector_type(4)));

#define B_    4
#define S_    2048
#define HID   2048
#define H_    16
#define HKV_  4
#define D_    128
#define M_    (B_*S_)
#define NQKV  ((H_+2*HKV_)*D_)   // 3072
#define QEND  (H_*D_)            // 2048
#define KEND  (QEND + HKV_*D_)   // 2560

__device__ __forceinline__ v16h cat16(v8h lo, v8h hi) {
  return __builtin_shufflevector(lo, hi, 0,1,2,3,4,5,6,7,8,9,10,11,12,13,14,15);
}
__device__ __forceinline__ v8f wmma32(v16h a, v16h b, v8f c) {
  return __builtin_amdgcn_wmma_f32_16x16x32_f16(false, a, false, b, (short)0, c, false, false);
}
__device__ __forceinline__ v8h cvt8(v4f a, v4f b) {
  v8h r;
  r[0]=(f16)a[0]; r[1]=(f16)a[1]; r[2]=(f16)a[2]; r[3]=(f16)a[3];
  r[4]=(f16)b[0]; r[5]=(f16)b[1]; r[6]=(f16)b[2]; r[7]=(f16)b[3];
  return r;
}

// ---------------------------------------------------------------------------
// Kernel 1: qkv = hidden @ w_qkv^T  (fp32 in, f16 WMMA w/ fp32 accum,
// f16 out scattered into head-major q16/k16/v16).
// Tile: 128(M) x 128(N) x 32(K). 8 waves as 4(M) x 2(N); wave = 32x64,
// 8 WMMA per k-step.
// ---------------------------------------------------------------------------
__global__ __launch_bounds__(256) void qkv_gemm_split(
    const float* __restrict__ Ah, const float* __restrict__ Wq,
    f16* __restrict__ q16, f16* __restrict__ k16, f16* __restrict__ v16)
{
  __shared__ __align__(16) f16 Alds[128*40];
  __shared__ __align__(16) f16 Wlds[128*40];
  const int tid  = threadIdx.x;
  const int lane = tid & 31, w = tid >> 5;
  const int wm = w & 3, wn = w >> 2;
  const int Mb = blockIdx.y * 128;
  const int Nb = blockIdx.x * 128;
  const int col16 = lane & 15;
  const int kbA  = (lane < 16) ? 0 : 16;
  const int kbB  = (lane < 16) ? 0 : 16;
  const int hi8  = (lane < 16) ? 0 : 8;

  const int srow = tid >> 1, sseg = (tid & 1) * 16;   // 128 rows x 32 k

  v8f acc[2][4] = {};

  for (int k0 = 0; k0 < HID; k0 += 32) {
    // stage A (f32 -> f16): 128x32
    const float* ap = Ah + (size_t)(Mb + srow) * HID + k0 + sseg;
    v4f a0 = *(const v4f*)(ap +  0);
    v4f a1 = *(const v4f*)(ap +  4);
    v4f a2 = *(const v4f*)(ap +  8);
    v4f a3 = *(const v4f*)(ap + 12);
    *(v8h*)&Alds[srow*40 + sseg    ] = cvt8(a0, a1);
    *(v8h*)&Alds[srow*40 + sseg + 8] = cvt8(a2, a3);
    // stage W (f32 -> f16): 128x32
    const float* wp = Wq + (size_t)(Nb + srow) * HID + k0 + sseg;
    v4f w0 = *(const v4f*)(wp +  0);
    v4f w1 = *(const v4f*)(wp +  4);
    v4f w2 = *(const v4f*)(wp +  8);
    v4f w3 = *(const v4f*)(wp + 12);
    *(v8h*)&Wlds[srow*40 + sseg    ] = cvt8(w0, w1);
    *(v8h*)&Wlds[srow*40 + sseg + 8] = cvt8(w2, w3);
    // prefetch next k-tile rows into cache while WMMAs run
    if (k0 + 32 < HID) {
      __builtin_prefetch(ap + 32, 0, 1);
      __builtin_prefetch(wp + 32, 0, 1);
    }
    __syncthreads();

    v16h af[2], bf[4];
#pragma unroll
    for (int i = 0; i < 2; ++i) {
      const f16* p = &Alds[(wm*32 + i*16 + col16)*40 + kbA];
      af[i] = cat16(*(const v8h*)p, *(const v8h*)(p + 8));
    }
#pragma unroll
    for (int j = 0; j < 4; ++j) {
      const f16* p = &Wlds[(wn*64 + j*16 + col16)*40 + kbB];
      bf[j] = cat16(*(const v8h*)p, *(const v8h*)(p + 8));
    }
#pragma unroll
    for (int i = 0; i < 2; ++i)
#pragma unroll
      for (int j = 0; j < 4; ++j)
        acc[i][j] = wmma32(af[i], bf[j], acc[i][j]);
    __syncthreads();
  }

  // epilogue: scatter to head-major f16 buffers (tile N-range stays within
  // one 128-wide head and one of the q/k/v regions)
#pragma unroll
  for (int i = 0; i < 2; ++i)
#pragma unroll
    for (int j = 0; j < 4; ++j) {
      const int n = Nb + wn*64 + j*16 + col16;
#pragma unroll
      for (int r = 0; r < 8; ++r) {
        const int m  = Mb + wm*32 + i*16 + r + hi8;
        const int bb = m >> 11, s = m & (S_-1);
        const f16 val = (f16)acc[i][j][r];
        if (n < QEND) {
          const int hh = n >> 7, d = n & 127;
          q16[(((size_t)bb*H_ + hh)*S_ + s)*D_ + d] = val;
        } else if (n < KEND) {
          const int hh = (n - QEND) >> 7, d = n & 127;
          k16[(((size_t)bb*HKV_ + hh)*S_ + s)*D_ + d] = val;
        } else {
          const int hh = (n - KEND) >> 7, d = n & 127;
          v16[(((size_t)bb*HKV_ + hh)*S_ + s)*D_ + d] = val;
        }
      }
    }
}

// ---------------------------------------------------------------------------
// Kernel 2: RoPE in place on q16 (with 1/sqrt(D) folded in) and k16.
// ---------------------------------------------------------------------------
__global__ __launch_bounds__(256) void rope_inplace(f16* __restrict__ q16,
                                                    f16* __restrict__ k16)
{
  const long long NQ = (long long)B_*H_*S_*64;
  const long long NK = (long long)B_*HKV_*S_*64;
  long long idx = (long long)blockIdx.x * 256 + threadIdx.x;
  f16* buf; float scl;
  if (idx < NQ) { buf = q16; scl = 0.08838834764831845f; }
  else {
    idx -= NQ;
    if (idx >= NK) return;
    buf = k16; scl = 1.0f;
  }
  const int d2 = (int)(idx & 63);
  const long long t = idx >> 6;
  const int s = (int)(t & (S_-1));
  const float freq = __expf(-(float)d2 * 0.14391156f);  // 10000^(-2*d2/128)
  float sn, cs;
  __sincosf((float)s * freq, &sn, &cs);
  f16* p = buf + t * D_;
  const float x1 = (float)p[d2];
  const float x2 = (float)p[d2 + 64];
  p[d2]      = (f16)((x1*cs - x2*sn) * scl);
  p[d2 + 64] = (f16)((x1*sn + x2*cs) * scl);
}

// ---------------------------------------------------------------------------
// Kernel 3: causal GQA flash attention. One WG per (b, h, 128-query tile),
// 8 waves x 16 queries. K tile async-staged to LDS, V staged transposed,
// streaming softmax, 16 WMMA per 32-key step.
// ---------------------------------------------------------------------------
__global__ __launch_bounds__(256) void attn_kernel(
    const f16* __restrict__ q16, const f16* __restrict__ k16,
    const f16* __restrict__ v16, f16* __restrict__ o16)
{
  __shared__ __align__(16) f16 Klds[32*136];    // 32 keys x 128 d (+pad)
  __shared__ __align__(16) f16 VTlds[128*40];   // transposed: d x 32 keys
  __shared__ __align__(16) f16 Plds[8*16*40];   // per-wave 16x32 P

  const int tid = threadIdx.x, lane = tid & 31, w = tid >> 5;
  const int bq = blockIdx.x * 128;
  const int h  = blockIdx.y, b = blockIdx.z, kh = h >> 2;
  const int qrow0 = bq + w*16;
  const int col16 = lane & 15;
  const bool hi = lane >= 16;
  const int hi8 = hi ? 8 : 0;
  const int kbA = hi ? 16 : 0;
  const int kbB = hi ? 16 : 0;

  const f16* qbase = q16 + (size_t)((b*H_  + h )*S_) * D_;
  const f16* kbase = k16 + (size_t)((b*HKV_+ kh)*S_) * D_;
  const f16* vbase = v16 + (size_t)((b*HKV_+ kh)*S_) * D_;

  // Q fragments in registers (16 rows x 128 d = 4 A-frags)
  v16h qf[4];
#pragma unroll
  for (int c = 0; c < 4; ++c) {
    const f16* qp = qbase + (size_t)(qrow0 + col16)*D_ + 32*c + kbA;
    qf[c] = cat16(*(const v8h*)qp, *(const v8h*)(qp + 8));
  }

  v8f o[8] = {};
  float mrow[8], lrow[8];
#pragma unroll
  for (int r = 0; r < 8; ++r) { mrow[r] = -1e30f; lrow[r] = 0.0f; }

  const int skey = tid >> 3, sseg = (tid & 7) * 16;
  const unsigned kldsoff =
      (unsigned)(uintptr_t)(&Klds[skey*136 + sseg]);
  const unsigned long long ksaddr = (unsigned long long)(uintptr_t)kbase;

  const int ktend = bq + 128;
  for (int kt = 0; kt < ktend; kt += 32) {
    // ---- async-stage K tile (f16, row-major, 2x16B per thread) ----
    const unsigned voff = (unsigned)(((kt + skey) * D_ + sseg) * 2);
    asm volatile("global_load_async_to_lds_b128 %0, %1, %2"
                 :: "v"(kldsoff), "v"(voff), "s"(ksaddr) : "memory");
    asm volatile("global_load_async_to_lds_b128 %0, %1, %2 offset:16"
                 :: "v"(kldsoff), "v"(voff), "s"(ksaddr) : "memory");
    // ---- stage V transposed ----
    const f16* vp = vbase + (size_t)(kt + skey) * D_ + sseg;
    v8h vv0 = *(const v8h*)vp;
    v8h vv1 = *(const v8h*)(vp + 8);
    if (kt + 32 < ktend) __builtin_prefetch(vp + 32*D_, 0, 1);
#pragma unroll
    for (int j = 0; j < 8; ++j) {
      VTlds[(sseg     + j)*40 + skey] = vv0[j];
      VTlds[(sseg + 8 + j)*40 + skey] = vv1[j];
    }
    asm volatile("s_wait_asynccnt 0" ::: "memory");
    __syncthreads();

    if (kt <= qrow0 + 15) {   // wave-uniform causal skip (EXEC stays all-1s)
      // ---- scores = Q @ K^T : two 16x16 frags over 32 keys ----
      v8f s0 = {}, s1 = {};
#pragma unroll
      for (int c = 0; c < 4; ++c) {
        const f16* kp0 = &Klds[(     col16)*136 + 32*c + kbB];
        const f16* kp1 = &Klds[(16 + col16)*136 + 32*c + kbB];
        s0 = wmma32(qf[c], cat16(*(const v8h*)kp0, *(const v8h*)(kp0+8)), s0);
        s1 = wmma32(qf[c], cat16(*(const v8h*)kp1, *(const v8h*)(kp1+8)), s1);
      }
      // ---- mask + streaming softmax (16-lane group reductions) ----
      float p0[8], p1[8], corr[8];
#pragma unroll
      for (int r = 0; r < 8; ++r) {
        const int row = qrow0 + r + hi8;
        const float x0 = (kt      + col16 > row) ? -3.0e38f : s0[r];
        const float x1 = (kt + 16 + col16 > row) ? -3.0e38f : s1[r];
        float tm = fmaxf(x0, x1);
        tm = fmaxf(tm, __shfl_xor(tm, 1, 32));
        tm = fmaxf(tm, __shfl_xor(tm, 2, 32));
        tm = fmaxf(tm, __shfl_xor(tm, 4, 32));
        tm = fmaxf(tm, __shfl_xor(tm, 8, 32));
        const float mn = fmaxf(mrow[r], tm);
        corr[r] = __expf(mrow[r] - mn);
        mrow[r] = mn;
        p0[r] = __expf(x0 - mn);
        p1[r] = __expf(x1 - mn);
        float rs = p0[r] + p1[r];
        rs += __shfl_xor(rs, 1, 32);
        rs += __shfl_xor(rs, 2, 32);
        rs += __shfl_xor(rs, 4, 32);
        rs += __shfl_xor(rs, 8, 32);
        lrow[r] = lrow[r]*corr[r] + rs;
      }
#pragma unroll
      for (int jd = 0; jd < 8; ++jd)
#pragma unroll
        for (int r = 0; r < 8; ++r) o[jd][r] *= corr[r];
      // ---- P: C-layout -> A-layout via per-wave LDS ----
      const int pb = w * 640;
#pragma unroll
      for (int r = 0; r < 8; ++r) {
        const int prow = r + hi8;
        Plds[pb + prow*40 +      col16] = (f16)p0[r];
        Plds[pb + prow*40 + 16 + col16] = (f16)p1[r];
      }
      const f16* pp = &Plds[pb + col16*40 + kbA];
      const v16h pa = cat16(*(const v8h*)pp, *(const v8h*)(pp + 8));
      // ---- out += P @ V ----
#pragma unroll
      for (int jd = 0; jd < 8; ++jd) {
        const f16* vq = &VTlds[(16*jd + col16)*40 + kbB];
        o[jd] = wmma32(pa, cat16(*(const v8h*)vq, *(const v8h*)(vq+8)), o[jd]);
      }
    }
    __syncthreads();
  }

  // epilogue: normalize and store o16[b][s][h][d]
#pragma unroll
  for (int r = 0; r < 8; ++r) {
    const float inv = 1.0f / lrow[r];
    const int srow = qrow0 + r + hi8;
    f16* op = o16 + ((size_t)(b*S_ + srow)*H_ + h)*D_;
#pragma unroll
    for (int jd = 0; jd < 8; ++jd)
      op[16*jd + col16] = (f16)(o[jd][r] * inv);
  }
}

// ---------------------------------------------------------------------------
// Kernel 4: out = attn_out @ w_o^T  (f16 A, f32 W converted in staging,
// fp32 output). Tile 128x128x32, 8 WMMA per wave per k-step.
// ---------------------------------------------------------------------------
__global__ __launch_bounds__(256) void out_gemm(
    const f16* __restrict__ Ao, const float* __restrict__ Wo,
    float* __restrict__ outp)
{
  __shared__ __align__(16) f16 Alds[128*40];
  __shared__ __align__(16) f16 Wlds[128*40];
  const int tid  = threadIdx.x;
  const int lane = tid & 31, w = tid >> 5;
  const int wm = w & 3, wn = w >> 2;
  const int Mb = blockIdx.y * 128;
  const int Nb = blockIdx.x * 128;
  const int col16 = lane & 15;
  const int kbA  = (lane < 16) ? 0 : 16;
  const int kbB  = (lane < 16) ? 0 : 16;
  const int hi8  = (lane < 16) ? 0 : 8;

  const int srow = tid >> 1, sseg = (tid & 1) * 16;

  v8f acc[2][4] = {};

  for (int k0 = 0; k0 < HID; k0 += 32) {
    // stage A (already f16): 128x32
    const f16* ap = Ao + (size_t)(Mb + srow) * HID + k0 + sseg;
    *(v8h*)&Alds[srow*40 + sseg    ] = *(const v8h*)(ap + 0);
    *(v8h*)&Alds[srow*40 + sseg + 8] = *(const v8h*)(ap + 8);
    // stage W (f32 -> f16): 128x32
    const float* wp = Wo + (size_t)(Nb + srow) * HID + k0 + sseg;
    v4f w0 = *(const v4f*)(wp +  0);
    v4f w1 = *(const v4f*)(wp +  4);
    v4f w2 = *(const v4f*)(wp +  8);
    v4f w3 = *(const v4f*)(wp + 12);
    *(v8h*)&Wlds[srow*40 + sseg    ] = cvt8(w0, w1);
    *(v8h*)&Wlds[srow*40 + sseg + 8] = cvt8(w2, w3);
    if (k0 + 32 < HID) {
      __builtin_prefetch(ap + 32, 0, 1);
      __builtin_prefetch(wp + 32, 0, 1);
    }
    __syncthreads();

    v16h af[2], bf[4];
#pragma unroll
    for (int i = 0; i < 2; ++i) {
      const f16* p = &Alds[(wm*32 + i*16 + col16)*40 + kbA];
      af[i] = cat16(*(const v8h*)p, *(const v8h*)(p + 8));
    }
#pragma unroll
    for (int j = 0; j < 4; ++j) {
      const f16* p = &Wlds[(wn*64 + j*16 + col16)*40 + kbB];
      bf[j] = cat16(*(const v8h*)p, *(const v8h*)(p + 8));
    }
#pragma unroll
    for (int i = 0; i < 2; ++i)
#pragma unroll
      for (int j = 0; j < 4; ++j)
        acc[i][j] = wmma32(af[i], bf[j], acc[i][j]);
    __syncthreads();
  }

#pragma unroll
  for (int i = 0; i < 2; ++i)
#pragma unroll
    for (int j = 0; j < 4; ++j) {
      const int n = Nb + wn*64 + j*16 + col16;
#pragma unroll
      for (int r = 0; r < 8; ++r) {
        const int m = Mb + wm*32 + i*16 + r + hi8;
        outp[(size_t)m * HID + n] = acc[i][j][r];
      }
    }
}

// ---------------------------------------------------------------------------
extern "C" void kernel_launch(void* const* d_in, const int* in_sizes, int n_in,
                              void* d_out, int out_size, void* d_ws, size_t ws_size,
                              hipStream_t stream) {
  const float* hidden = (const float*)d_in[0];
  const float* w_qkv  = (const float*)d_in[1];
  const float* w_o    = (const float*)d_in[2];
  float* outp = (float*)d_out;

  char* ws = (char*)d_ws;
  f16* q16 = (f16*)(ws + 0);                    // 32 MB
  f16* k16 = (f16*)(ws + (size_t)33554432);     //  8 MB
  f16* v16 = (f16*)(ws + (size_t)41943040);     //  8 MB
  f16* o16 = (f16*)(ws + (size_t)50331648);     // 32 MB

  dim3 g1(NQKV/128, M_/128);         // 24 x 64
  qkv_gemm_split<<<g1, 256, 0, stream>>>(hidden, w_qkv, q16, k16, v16);

  const long long nrope = ((long long)B_*H_*S_ + (long long)B_*HKV_*S_) * 64;
  rope_inplace<<<(int)((nrope + 255) / 256), 256, 0, stream>>>(q16, k16);

  dim3 g3(S_/128, H_, B_);           // 16 x 16 x 4
  attn_kernel<<<g3, 256, 0, stream>>>(q16, k16, v16, o16);

  dim3 g4(HID/128, M_/128);          // 16 x 64
  out_gemm<<<g4, 256, 0, stream>>>(o16, w_o, outp);
}
